// Quantizer_87393994539746
// MI455X (gfx1250) — matrix operand order repaired and checked by reading, not compiled, not run on hardware
//
#include <hip/hip_runtime.h>
#include <stdint.h>

typedef __attribute__((ext_vector_type(16))) _Float16 v16h;
typedef __attribute__((ext_vector_type(8)))  _Float16 v8h;
typedef __attribute__((ext_vector_type(8)))  float    v8f;
typedef __attribute__((ext_vector_type(4)))  float    v4f;

#define DDIM 256
#define KCODES 4096
#define CHUNK 64                         // codes staged in LDS per iteration
#define NCHUNKS (KCODES / CHUNK)         // 64
#define ROWS_PER_WAVE 16
#define WAVES_PER_WG 8
#define ROWS_PER_WG (ROWS_PER_WAVE * WAVES_PER_WG)   // 128
#define LDS_STRIDE 264                   // halves per row: 256 + 8 pad (breaks bank conflicts)

// ---------------- kernel 1: b2[k] = ||c_k||^2  AND  f16 copy of codebook ----------------
__global__ void vq_prep_kernel(const float* __restrict__ cb,
                               float* __restrict__ b2,
                               _Float16* __restrict__ cbh) {
    const int t = blockIdx.x * blockDim.x + threadIdx.x;   // 0 .. KCODES*DDIM/4 - 1
    // convert 4 contiguous elements (coalesced float4 read / b64 write)
    {
        v4f v = *(const v4f*)(cb + (size_t)t * 4);
        _Float16* d = cbh + (size_t)t * 4;
        d[0] = (_Float16)v.x; d[1] = (_Float16)v.y;
        d[2] = (_Float16)v.z; d[3] = (_Float16)v.w;
    }
    // first 4096 threads also produce b2 (tiny: 4 MB total read, L2-resident)
    if (t < KCODES) {
        const float* row = cb + (size_t)t * DDIM;
        float s = 0.f;
#pragma unroll 4
        for (int i = 0; i < DDIM / 4; ++i) {
            v4f v = *(const v4f*)(row + 4 * i);
            s += v.x * v.x + v.y * v.y + v.z * v.z + v.w * v.w;
        }
        b2[t] = s;
    }
}

// ---- CDNA5 async global->LDS copy (ASYNCcnt path), GVS addressing ----
__device__ __forceinline__ void async_copy_b128(const _Float16* gbase,
                                                uint32_t goff_bytes,
                                                uint32_t lds_off_bytes) {
    asm volatile("global_load_async_to_lds_b128 %0, %1, %2"
                 :: "v"(lds_off_bytes), "v"(goff_bytes), "s"(gbase)
                 : "memory");
}
__device__ __forceinline__ void wait_asynccnt0() {
    asm volatile("s_wait_asynccnt 0" ::: "memory");
}

// stage one 64-code x 256-dim f16 chunk (32 KB) into LDS: 8 x b128 per thread
__device__ __forceinline__ void stage_chunk_async(const _Float16* cbh, int cbase,
                                                  uint32_t lds_base, int tid) {
#pragma unroll
    for (int i = 0; i < 8; ++i) {
        const int g    = i * 256 + tid;      // 16-byte granule id, 0..2047
        const int code = g >> 5;             // 32 granules per 512-byte row
        const int sub  = g & 31;
        const uint32_t goff = (uint32_t)(cbase + code) * (DDIM * 2u) + (uint32_t)sub * 16u;
        const uint32_t loff = lds_base + (uint32_t)code * (LDS_STRIDE * 2u) + (uint32_t)sub * 16u;
        async_copy_b128(cbh, goff, loff);
    }
}

// ---------------- kernel 2: fused distance-GEMM + argmin + gather ----------------
__launch_bounds__(256)
__global__ void vq_main_kernel(const float* __restrict__ x,
                               const float* __restrict__ cb,
                               const float* __restrict__ b2,
                               const _Float16* __restrict__ cbh,
                               float* __restrict__ out) {
    __shared__ __align__(16) _Float16 Bs[2][CHUNK][LDS_STRIDE];

    const int tid    = threadIdx.x;
    const int wave   = tid >> 5;
    const int lane   = tid & 31;
    const int lane16 = lane & 15;
    const int half   = lane >> 4;

    const int row_base = blockIdx.x * ROWS_PER_WG + wave * ROWS_PER_WAVE;

    const uint32_t lds0 = (uint32_t)(uintptr_t)&Bs[0][0][0];
    const uint32_t lds1 = (uint32_t)(uintptr_t)&Bs[1][0][0];

    // prefetch chunk 0 into buffer 0 while we build the A panel
    stage_chunk_async(cbh, 0, lds0, tid);

    // ---- A panel: 16 rows x 256 dims, f32 -> f16, resident in VGPRs for whole kernel ----
    // ISA A layout (16-bit, 16x32): lane m = lane%16 holds row m; lanes<16 cover
    // K = d0..d0+7 (elems 0-7) and d0+16..d0+23 (elems 8-15); lanes>=16 are +8.
    const float* xrow = x + (size_t)(row_base + lane16) * DDIM;
    v16h afrag[8];
#pragma unroll
    for (int s = 0; s < 8; ++s) {
        const int d0 = 32 * s + 8 * half;
        v4f p0 = *(const v4f*)(xrow + d0);
        v4f p1 = *(const v4f*)(xrow + d0 + 4);
        v4f p2 = *(const v4f*)(xrow + d0 + 16);
        v4f p3 = *(const v4f*)(xrow + d0 + 20);
        v16h a;
        a[0]  = (_Float16)p0.x; a[1]  = (_Float16)p0.y; a[2]  = (_Float16)p0.z; a[3]  = (_Float16)p0.w;
        a[4]  = (_Float16)p1.x; a[5]  = (_Float16)p1.y; a[6]  = (_Float16)p1.z; a[7]  = (_Float16)p1.w;
        a[8]  = (_Float16)p2.x; a[9]  = (_Float16)p2.y; a[10] = (_Float16)p2.z; a[11] = (_Float16)p2.w;
        a[12] = (_Float16)p3.x; a[13] = (_Float16)p3.y; a[14] = (_Float16)p3.z; a[15] = (_Float16)p3.w;
        afrag[s] = a;
    }

    // running (min, idx) per C-VGPR slot: VGPR v holds row  v + 8*half
    float rmin[8];
    int   ridx[8];
#pragma unroll
    for (int v = 0; v < 8; ++v) { rmin[v] = __builtin_inff(); ridx[v] = 0; }

    for (int c = 0; c < NCHUNKS; ++c) {
        const int cbase = c * CHUNK;

        wait_asynccnt0();       // my async issues have landed in LDS
        __syncthreads();        // everyone's have; everyone done computing on the other buffer

        // kick off next chunk into the other buffer (overlaps with WMMA below)
        if (c + 1 < NCHUNKS)
            stage_chunk_async(cbh, cbase + CHUNK, (c & 1) ? lds0 : lds1, tid);

        const _Float16 (*B)[LDS_STRIDE] = Bs[c & 1];

        v8f acc[4];
#pragma unroll
        for (int j = 0; j < 4; ++j) acc[j] = (v8f){0.f, 0.f, 0.f, 0.f, 0.f, 0.f, 0.f, 0.f};

#pragma unroll
        for (int s = 0; s < 8; ++s) {
            const int d0 = 32 * s + 8 * half;    // B layout: lane n, elems 0-7 -> K=d0..d0+7, elems 8-15 -> K=d0+16..d0+23
#pragma unroll
            for (int j = 0; j < 4; ++j) {
                const int code = j * 16 + lane16;
                v8h b0 = *(const v8h*)&B[code][d0];
                v8h b1 = *(const v8h*)&B[code][d0 + 16];
                v16h b;
#pragma unroll
                for (int e = 0; e < 8; ++e) { b[e] = b0[e]; b[8 + e] = b1[e]; }
                acc[j] = __builtin_amdgcn_wmma_f32_16x16x32_f16(
                    false, afrag[s], false, b, (short)0, acc[j], false, false);
            }
        }

        // fused argmin over this 64-code chunk
        float b2v[4];
#pragma unroll
        for (int j = 0; j < 4; ++j) b2v[j] = b2[cbase + j * 16 + lane16];

#pragma unroll
        for (int v = 0; v < 8; ++v) {
            float bestv = b2v[0] - 2.0f * acc[0][v];
            int   besti = cbase + lane16;
#pragma unroll
            for (int j = 1; j < 4; ++j) {
                float val = b2v[j] - 2.0f * acc[j][v];
                int   idx = cbase + j * 16 + lane16;
                if (val < bestv || (val == bestv && idx < besti)) { bestv = val; besti = idx; }
            }
            // butterfly across the 16 lanes of this half (xor < 16 stays in-half)
#pragma unroll
            for (int off = 1; off < 16; off <<= 1) {
                float ov = __shfl_xor(bestv, off, 32);
                int   oi = __shfl_xor(besti, off, 32);
                if (ov < bestv || (ov == bestv && oi < besti)) { bestv = ov; besti = oi; }
            }
            if (bestv < rmin[v] || (bestv == rmin[v] && besti < ridx[v])) {
                rmin[v] = bestv; ridx[v] = besti;
            }
        }
    }

    // gather: out[row] = codebook_f32[best]; 16 lanes of each half copy one 1KB row
#pragma unroll
    for (int v = 0; v < 8; ++v) {
        const int row = row_base + v + 8 * half;
        const float* src = cb + (size_t)ridx[v] * DDIM;
        float*       dst = out + (size_t)row * DDIM;
#pragma unroll
        for (int i = 0; i < 4; ++i) {
            const int d = lane16 * 16 + i * 4;
            *(v4f*)(dst + d) = *(const v4f*)(src + d);
        }
    }
}

extern "C" void kernel_launch(void* const* d_in, const int* in_sizes, int n_in,
                              void* d_out, int out_size, void* d_ws, size_t ws_size,
                              hipStream_t stream) {
    const float* x  = (const float*)d_in[0];   // [N, 256]
    const float* cb = (const float*)d_in[1];   // [4096, 256]
    float* out = (float*)d_out;                // [N, 256]

    // workspace layout: [0,16KB) = b2 (4096 f32); [16KB, 16KB+2MB) = f16 codebook
    float*    b2  = (float*)d_ws;
    _Float16* cbh = (_Float16*)((char*)d_ws + KCODES * sizeof(float));

    const int N = in_sizes[0] / DDIM;          // 65536

    vq_prep_kernel<<<(KCODES * DDIM / 4) / 256, 256, 0, stream>>>(cb, b2, cbh);
    vq_main_kernel<<<N / ROWS_PER_WG, 256, 0, stream>>>(x, cb, b2, cbh, out);
}
